// KernelRegressor_65481071409470
// MI455X (gfx1250) — matrix operand order repaired and checked by reading, not compiled, not run on hardware
//
#include <hip/hip_runtime.h>
#include <hip/hip_bf16.h>

typedef __attribute__((ext_vector_type(16))) _Float16 v16h;
typedef __attribute__((ext_vector_type(8)))  _Float16 v8h;
typedef __attribute__((ext_vector_type(8)))  float    v8f;
typedef __attribute__((ext_vector_type(4)))  int      v4i;

#define N_TEST  8192
#define N_TRAIN 8192
#define DIM     512
#define NOUT    16
#define GAMMA   0.001f
#define NSPLIT  8
#define NSLICE  (N_TRAIN / NSPLIT)

#define TPAD 520     // LDS row stride (halves) for train tile: 1040B = 260 dw -> 4-bank row shift, conflict-free
#define KPAD 40      // LDS row stride (halves) for K-transpose tile: 80B = 20 dw -> conflict-free

#define GPTR __attribute__((address_space(1)))
#define LPTR __attribute__((address_space(3)))

#if __has_builtin(__builtin_amdgcn_global_load_async_to_lds_b128)
#define ASYNC_B128(g, l) \
    __builtin_amdgcn_global_load_async_to_lds_b128((GPTR v4i*)(g), (LPTR v4i*)(l), 0, 0)
#else
#define ASYNC_B128(g, l) \
    asm volatile("global_load_async_to_lds_b128 %0, %1, off" \
                 :: "v"((unsigned)(size_t)(l)), "v"((const void*)(g)) : "memory")
#endif

#if __has_builtin(__builtin_amdgcn_s_wait_asynccnt)
#define WAIT_ASYNC0() __builtin_amdgcn_s_wait_asynccnt(0)
#else
#define WAIT_ASYNC0() asm volatile("s_wait_asynccnt 0" ::: "memory")
#endif

// ---------------------------------------------------------------------------
// prep_rows: one block per row (rows 0..8191 = X, 8192..16383 = train_X).
// f32 -> f16 conversion + row sum-of-squares.
// ---------------------------------------------------------------------------
__global__ __launch_bounds__(256)
void prep_rows(const float* __restrict__ X, const float* __restrict__ T,
               _Float16* __restrict__ Xh, _Float16* __restrict__ Th,
               float* __restrict__ xsq, float* __restrict__ ysq)
{
    __shared__ float red[256];
    const int row  = blockIdx.x;
    const bool isX = row < N_TEST;
    const int r    = isX ? row : row - N_TEST;
    const float*    src = (isX ? X  : T)  + (size_t)r * DIM;
    _Float16*       dst = (isX ? Xh : Th) + (size_t)r * DIM;
    const int t = threadIdx.x;

    const float v0 = src[t];
    const float v1 = src[t + 256];
    dst[t]       = (_Float16)v0;
    dst[t + 256] = (_Float16)v1;

    red[t] = v0 * v0 + v1 * v1;
    __syncthreads();
    #pragma unroll
    for (int s = 128; s > 0; s >>= 1) {
        if (t < s) red[t] += red[t + s];
        __syncthreads();
    }
    if (t == 0) (isX ? xsq : ysq)[r] = red[0];
}

// ---------------------------------------------------------------------------
// prep_alpha: alpha_T[16][8192] in f16; zero d_out (harness poisons it).
// ---------------------------------------------------------------------------
__global__ __launch_bounds__(256)
void prep_alpha(const float* __restrict__ alpha, _Float16* __restrict__ alphaT,
                float* __restrict__ out)
{
    const int n = blockIdx.x * 256 + threadIdx.x;
    #pragma unroll
    for (int o = 0; o < NOUT; ++o) {
        alphaT[(size_t)o * N_TRAIN + n] = (_Float16)alpha[(size_t)n * NOUT + o];
        out[(size_t)n * NOUT + o] = 0.0f;
    }
}

// ---------------------------------------------------------------------------
// helpers for the fused kernel
// ---------------------------------------------------------------------------
__device__ __forceinline__ v16h combine16(v8h lo, v8h hi)
{
    v16h r;
    #pragma unroll
    for (int i = 0; i < 8; ++i) { r[i] = lo[i]; r[8 + i] = hi[i]; }
    return r;
}

// B fragment (32x16): lane = column n, two 8-half K(=dim) runs (sub*8, 16+sub*8) from LDS
__device__ __forceinline__ v16h lds_bfrag(const _Float16* base, int row, int k, int sub)
{
    const _Float16* p = base + row * TPAD + k + sub * 8;
    return combine16(*(const v8h*)(p), *(const v8h*)(p + 16));
}

// async copy of 32 rows x 1024B from global (contiguous rows) into padded LDS tile
__device__ __forceinline__ void async_fill_tile(const _Float16* gsrc, _Float16* lbase, int tid)
{
    const int row = tid >> 3;          // 32 rows, 8 threads per row
    const int c   = tid & 7;
    const char* g = (const char*)gsrc + row * 1024 + c * 16;
    char*       l = (char*)lbase      + row * (TPAD * 2) + c * 16;
    #pragma unroll
    for (int j = 0; j < 8; ++j)
        ASYNC_B128(g + j * 128, l + j * 128);
}

// ---------------------------------------------------------------------------
// rbf_fused: grid (64, 8), 256 threads = 8 waves. Wave w owns 16 test rows.
// The block's 32-row train tile is async-staged into double-buffered LDS;
// cross GEMM reads B from LDS, exp epilogue, per-wave LDS transpose, then
// one WMMA vs alpha_T accumulating the 16x16 (M x OUT) tile.
// ---------------------------------------------------------------------------
__global__ __launch_bounds__(256)
void rbf_fused(const _Float16* __restrict__ Xh, const _Float16* __restrict__ Th,
               const _Float16* __restrict__ alphaT,
               const float* __restrict__ xsq, const float* __restrict__ ysq,
               float* __restrict__ out)
{
    __shared__ __align__(16) _Float16 ldsT[2][32 * TPAD];   // ~65 KB double-buffered train tile
    __shared__ __align__(16) _Float16 ldsK[8][16 * KPAD];   // ~10 KB per-wave transpose scratch

    const int tid  = threadIdx.x;
    const int lane = tid & 31;
    const int wave = tid >> 5;
    const int sub  = lane >> 4;
    const int l15  = lane & 15;
    const int m_base = blockIdx.x * 128 + wave * 16;
    const int n0     = blockIdx.y * NSLICE;
    _Float16* myK = &ldsK[wave][0];

    // kick off the first train tile before anything else
    async_fill_tile(Th + (size_t)n0 * DIM, &ldsT[0][0], tid);

    // hoist this wave's X A-fragments (16x512 f16) into registers for the
    // entire kernel: A layout = two contiguous 8-half runs per lane
    const _Float16* arow = Xh + (size_t)(m_base + l15) * DIM + sub * 8;
    v16h af[16];
    #pragma unroll
    for (int kk = 0; kk < 16; ++kk)
        af[kk] = combine16(*(const v8h*)(arow + kk * 32),
                           *(const v8h*)(arow + kk * 32 + 16));

    float xs[8];
    #pragma unroll
    for (int i = 0; i < 8; ++i) xs[i] = xsq[m_base + i + sub * 8];

    v8f oacc = {};

    WAIT_ASYNC0();
    __syncthreads();

    int cur = 0;
    for (int n = n0; n < n0 + NSLICE; n += 32) {
        // stream the next tile into the other buffer while we compute
        if (n + 32 < n0 + NSLICE)
            async_fill_tile(Th + (size_t)(n + 32) * DIM, &ldsT[cur ^ 1][0], tid);

        const _Float16* bbuf = &ldsT[cur][0];
        v8f c0 = {}, c1 = {};

        // software-pipelined k-loop: B fragments one step ahead in registers
        v16h b0 = lds_bfrag(bbuf, l15,      0, sub);
        v16h b1 = lds_bfrag(bbuf, l15 + 16, 0, sub);
        #pragma unroll
        for (int kk = 0; kk < 16; ++kk) {
            v16h nb0 = b0, nb1 = b1;
            if (kk < 15) {
                nb0 = lds_bfrag(bbuf, l15,      (kk + 1) * 32, sub);
                nb1 = lds_bfrag(bbuf, l15 + 16, (kk + 1) * 32, sub);
            }
            c0 = __builtin_amdgcn_wmma_f32_16x16x32_f16(false, af[kk], false, b0,
                                                        (short)0, c0, false, false);
            c1 = __builtin_amdgcn_wmma_f32_16x16x32_f16(false, af[kk], false, b1,
                                                        (short)0, c1, false, false);
            b0 = nb0; b1 = nb1;
        }

        // RBF epilogue; store K tile f16 into per-wave LDS (DS is in-order
        // within a wave, so no barrier is needed around the transpose)
        const float ys0 = ysq[n + l15];
        const float ys1 = ysq[n + 16 + l15];
        #pragma unroll
        for (int i = 0; i < 8; ++i) {
            const int m = i + sub * 8;
            const float k0 = __expf(-GAMMA * (xs[i] + ys0 - 2.0f * c0[i]));
            const float k1 = __expf(-GAMMA * (xs[i] + ys1 - 2.0f * c1[i]));
            myK[m * KPAD + l15]      = (_Float16)k0;
            myK[m * KPAD + 16 + l15] = (_Float16)k1;
        }

        // re-read K tile as an A fragment (transpose through LDS)
        const _Float16* krow = myK + l15 * KPAD + sub * 8;
        v16h aK = combine16(*(const v8h*)(krow), *(const v8h*)(krow + 16));

        // alpha B fragment: column = output index, two 8-value n runs (sub*8, 16+sub*8)
        const _Float16* ap = alphaT + (size_t)l15 * N_TRAIN + n + sub * 8;
        v16h bA = combine16(*(const v8h*)(ap), *(const v8h*)(ap + 16));

        oacc = __builtin_amdgcn_wmma_f32_16x16x32_f16(false, aK, false, bA,
                                                      (short)0, oacc, false, false);

        WAIT_ASYNC0();          // next tile landed (own ops)
        __syncthreads();        // all waves done reading cur + filling next
        cur ^= 1;
    }

    #pragma unroll
    for (int i = 0; i < 8; ++i)
        atomicAdd(&out[(size_t)(m_base + i + sub * 8) * NOUT + l15], oacc[i]);
}

// ---------------------------------------------------------------------------
extern "C" void kernel_launch(void* const* d_in, const int* in_sizes, int n_in,
                              void* d_out, int out_size, void* d_ws, size_t ws_size,
                              hipStream_t stream)
{
    const float* X     = (const float*)d_in[0];
    const float* trX   = (const float*)d_in[1];
    const float* alpha = (const float*)d_in[2];
    float* out = (float*)d_out;

    char* ws = (char*)d_ws;
    _Float16* Xh     = (_Float16*)(ws);
    _Float16* Th     = (_Float16*)(ws + (size_t)N_TEST * DIM * 2);
    _Float16* alphaT = (_Float16*)(ws + (size_t)(N_TEST + N_TRAIN) * DIM * 2);
    float*    xsq    = (float*)((char*)alphaT + (size_t)NOUT * N_TRAIN * 2);
    float*    ysq    = xsq + N_TEST;

    prep_rows<<<N_TEST + N_TRAIN, 256, 0, stream>>>(X, trX, Xh, Th, xsq, ysq);
    prep_alpha<<<N_TRAIN / 256, 256, 0, stream>>>(alpha, alphaT, out);
    rbf_fused<<<dim3(N_TEST / 128, NSPLIT), 256, 0, stream>>>(Xh, Th, alphaT,
                                                              xsq, ysq, out);
}
